// SpatialTransformer_41592463294583
// MI455X (gfx1250) — matrix operand order repaired and checked by reading, not compiled, not run on hardware
//
#include <hip/hip_runtime.h>
#include <hip/hip_bf16.h>

// ---------------------------------------------------------------------------
// SpatialTransformer for MI455X (gfx1250), wave32 + v_wmma_f32_16x16x32_f16.
//
// D=256, H=4, HD=64, NL=4, SEQ=30, IMG=16384, B=128.
// preds[t] (the per-step state) lives in ws; each step is a chain of small
// WMMA GEMM kernels (latency-bound recurrence), encoder/decoder are big
// WMMA GEMMs (decoder is bandwidth-bound on the 251 MB output write).
// Epilogue behavior (residual / relu / layernorm / decoder-remap) is a
// compile-time template so the inner loops are branch-free.
// ---------------------------------------------------------------------------

typedef __attribute__((ext_vector_type(16))) _Float16 v16h;
typedef __attribute__((ext_vector_type(8)))  float    v8f;

#define DD   256
#define BB   128
#define IMG_ 16384
#define SEQ_ 30
#define NL_  4

static __device__ __forceinline__ v8f wmma_f16(v16h a, v16h b, v8f c) {
    return __builtin_amdgcn_wmma_f32_16x16x32_f16(
        /*neg_a=*/false, a, /*neg_b=*/false, b,
        /*c_mod=*/(short)0, c, /*reuse_a=*/false, /*reuse_b=*/false);
}

// Build one K=32 slice of the A fragment (two contiguous 8-float runs).
static __device__ __forceinline__ v16h load_a_frag(const float* __restrict__ Ar,
                                                   int k0, int lh) {
    v16h a;
    const int kb = k0 + lh * 16;
    #pragma unroll
    for (int i = 0; i < 8; ++i) a[i]     = (_Float16)Ar[kb + i];
    #pragma unroll
    for (int i = 0; i < 8; ++i) a[8 + i] = (_Float16)Ar[kb + 8 + i];
    return a;
}

// ---------------------------------------------------------------------------
// Generic GEMM: out = f(A[M,K] @ W[K,N] + bias [+ residual]) with optional
// ReLU or row-LayerNorm (LN requires N==256 and gridDim.y==1 so a full row
// lives in one block's LDS stripe).
//
// Block = 256 threads = 8 waves. blockIdx.x selects a 16-row stripe.
// Wave w handles column tiles  ct = blockIdx.y*8 + w, step gridDim.y*8.
// DEC remaps output rows (row = t*128+b) to  b*SEQ*IMG + t*IMG.
// ---------------------------------------------------------------------------
template <bool RES, bool RELU, bool LN, bool DEC>
__global__ void k_gemm(const float* __restrict__ A, long lda,
                       const float* __restrict__ W, long ldw,
                       const float* __restrict__ bias,
                       const float* __restrict__ residual, long ldres,
                       const float* __restrict__ ln_g, const float* __restrict__ ln_b,
                       float* __restrict__ out,
                       int N, int K) {
    __shared__ float stripe[16 * 256];          // only used on the LN path
    __shared__ float s_mu[16], s_rs[16];

    const int tid  = threadIdx.x;
    const int wave = tid >> 5;
    const int lane = tid & 31;
    const int m    = lane & 15;                 // row within tile (A) / col (B,C)
    const int lh   = lane >> 4;                 // lane-half selector
    const int rowBase = blockIdx.x * 16;

    const int numColTiles = N >> 4;
    const int ctStep      = gridDim.y * 8;

    for (int ct = blockIdx.y * 8 + wave; ct < numColTiles; ct += ctStep) {
        const int colBase = ct << 4;
        const int col     = colBase + m;
        v8f c = {};

        const float* Ar = A + (size_t)(rowBase + m) * lda;
        const float* Wc = W + col;
        #pragma unroll 2
        for (int k0 = 0; k0 < K; k0 += 32) {
            v16h a = load_a_frag(Ar, k0, lh);
            v16h b;   // lane holds column `col`; K = k0 + lh*16 + i
            const float* Wk = Wc + (size_t)(k0 + lh * 16) * ldw;
            #pragma unroll
            for (int i = 0; i < 16; ++i) b[i] = (_Float16)Wk[(size_t)i * ldw];
            c = wmma_f16(a, b, c);
        }

        const float bv = bias[col];
        // C/D layout: lane -> column col ; VGPR r -> row rowBase + r + lh*8
        #pragma unroll
        for (int r = 0; r < 8; ++r) {
            const int row = rowBase + r + lh * 8;
            float v = c[r] + bv;
            if (RES)  v += residual[(size_t)row * ldres + col];
            if (RELU) v = fmaxf(v, 0.0f);
            if (LN) {
                stripe[(r + lh * 8) * 256 + col] = v;   // N==256 on LN path
            } else if (DEC) {
                out[(size_t)(row & 127) * ((size_t)SEQ_ * IMG_) +
                    (size_t)(row >> 7) * IMG_ + col] = v;
            } else {
                out[(size_t)row * N + col] = v;
            }
        }
    }

    if (LN) {                                    // fused row LayerNorm, N==256
        __syncthreads();
        if (tid < 16) {
            float s = 0.f, s2 = 0.f;
            const float* rp = &stripe[tid * 256];
            for (int j = 0; j < 256; ++j) { float v = rp[j]; s += v; s2 += v * v; }
            const float mu  = s * (1.0f / 256.0f);
            const float var = s2 * (1.0f / 256.0f) - mu * mu;
            s_mu[tid] = mu;
            s_rs[tid] = rsqrtf(var + 1e-5f);
        }
        __syncthreads();
        const int r  = tid >> 4;
        const int c0 = tid & 15;
        const float mu = s_mu[r], rs = s_rs[r];
        const int row  = rowBase + r;
        #pragma unroll
        for (int j = 0; j < 16; ++j) {
            const int col = c0 + j * 16;
            out[(size_t)row * 256 + col] =
                (stripe[r * 256 + col] - mu) * rs * ln_g[col] + ln_b[col];
        }
    }
}

// ---------------------------------------------------------------------------
// Attention for one layer: 4 blocks (one per head), 512 threads = 16 waves.
// qkv: [128, 768] fp32 (q|k|v).  sc: [4,128,128] fp32 scratch.
// out: [128, 256] fp32 (heads concatenated).
// K (then V) staged in LDS as f16; scores + probs@V via WMMA.
// ---------------------------------------------------------------------------
__global__ void k_attn(const float* __restrict__ qkv,
                       float* __restrict__ sc,
                       float* __restrict__ out) {
    __shared__ _Float16 kv[128 * 64];            // K, then reused for V
    const int h    = blockIdx.x;
    const int tid  = threadIdx.x;
    const int wave = tid >> 5;
    const int lane = tid & 31;
    const int m    = lane & 15;
    const int lh   = lane >> 4;
    float* sch = sc + (size_t)h * 128 * 128;

    // stage K (f16)
    for (int i = tid; i < 128 * 64; i += 512) {
        const int r = i >> 6, c = i & 63;
        kv[i] = (_Float16)qkv[(size_t)r * 768 + 256 + h * 64 + c];
    }
    __syncthreads();

    // scores = (q * 1/8) @ K^T : 8x8 tiles, 16 waves -> 4 tiles each
    for (int tile = wave; tile < 64; tile += 16) {
        const int rowBase = (tile >> 3) << 4;
        const int colBase = (tile & 7) << 4;
        v8f c = {};
        const float* Qr = qkv + (size_t)(rowBase + m) * 768 + h * 64;
        #pragma unroll
        for (int k0 = 0; k0 < 64; k0 += 32) {
            v16h a;
            const int kb = k0 + lh * 16;
            #pragma unroll
            for (int i = 0; i < 8; ++i) a[i]     = (_Float16)(Qr[kb + i]      * 0.125f);
            #pragma unroll
            for (int i = 0; i < 8; ++i) a[8 + i] = (_Float16)(Qr[kb + 8 + i] * 0.125f);
            v16h b;   // B[k,n] = K[colBase+n, k]  (transposed read from LDS)
            const int kk = k0 + lh * 16;
            const _Float16* Kp = &kv[(colBase + m) * 64 + kk];
            #pragma unroll
            for (int i = 0; i < 16; ++i) b[i] = Kp[i];
            c = wmma_f16(a, b, c);
        }
        #pragma unroll
        for (int r = 0; r < 8; ++r)
            sch[(size_t)(rowBase + r + lh * 8) * 128 + colBase + m] = c[r];
    }
    __threadfence();
    __syncthreads();

    // softmax: one thread per row
    if (tid < 128) {
        float* srow = sch + (size_t)tid * 128;
        float mx = -3.402823e38f;
        for (int j = 0; j < 128; ++j) mx = fmaxf(mx, srow[j]);
        float s = 0.f;
        for (int j = 0; j < 128; ++j) { float e = __expf(srow[j] - mx); srow[j] = e; s += e; }
        const float inv = 1.0f / s;
        for (int j = 0; j < 128; ++j) srow[j] *= inv;
    }
    __threadfence();
    __syncthreads();

    // stage V (f16), reusing kv
    for (int i = tid; i < 128 * 64; i += 512) {
        const int r = i >> 6, c = i & 63;
        kv[i] = (_Float16)qkv[(size_t)r * 768 + 512 + h * 64 + c];
    }
    __syncthreads();

    // O = probs @ V : 8x4 tiles, 16 waves -> 2 tiles each
    for (int tile = wave; tile < 32; tile += 16) {
        const int rowBase = (tile >> 2) << 4;
        const int colBase = (tile & 3) << 4;
        v8f c = {};
        const float* Pr = sch + (size_t)(rowBase + m) * 128;
        #pragma unroll
        for (int k0 = 0; k0 < 128; k0 += 32) {
            v16h a = load_a_frag(Pr, k0, lh);
            v16h b;   // B[k,n] = V[k, n]
            const int kk = k0 + lh * 16;
            const _Float16* Vp = &kv[kk * 64 + colBase + m];
            #pragma unroll
            for (int i = 0; i < 16; ++i) b[i] = Vp[i * 64];
            c = wmma_f16(a, b, c);
        }
        #pragma unroll
        for (int r = 0; r < 8; ++r)
            out[(size_t)(rowBase + r + lh * 8) * 256 + h * 64 + colBase + m] = c[r];
    }
}

// ---------------------------------------------------------------------------
// Standalone LayerNorm over rows of 256 (decn after each step).
// grid = #rows, 256 threads.
// ---------------------------------------------------------------------------
__global__ void k_layernorm(const float* __restrict__ in,
                            const float* __restrict__ g,
                            const float* __restrict__ b,
                            float* __restrict__ out) {
    __shared__ float red[256];
    const int tid = threadIdx.x;
    const size_t base = (size_t)blockIdx.x * 256;
    const float v = in[base + tid];

    red[tid] = v; __syncthreads();
    for (int s = 128; s > 0; s >>= 1) { if (tid < s) red[tid] += red[tid + s]; __syncthreads(); }
    const float mu = red[0] * (1.0f / 256.0f);
    __syncthreads();
    const float d = v - mu;
    red[tid] = d * d; __syncthreads();
    for (int s = 128; s > 0; s >>= 1) { if (tid < s) red[tid] += red[tid + s]; __syncthreads(); }
    const float rs = rsqrtf(red[0] * (1.0f / 256.0f) + 1e-5f);
    out[base + tid] = d * rs * g[tid] + b[tid];
}

// ---------------------------------------------------------------------------
extern "C" void kernel_launch(void* const* d_in, const int* in_sizes, int n_in,
                              void* d_out, int out_size, void* d_ws, size_t ws_size,
                              hipStream_t stream) {
    const float* x      = (const float*)d_in[0];
    const float* enc_W  = (const float*)d_in[1];
    const float* enc_b  = (const float*)d_in[2];
    const float* encn_g = (const float*)d_in[3];
    const float* encn_b = (const float*)d_in[4];
    const float* dec_W  = (const float*)d_in[5];
    const float* dec_b  = (const float*)d_in[6];
    const float* decn_g = (const float*)d_in[7];
    const float* decn_b = (const float*)d_in[8];
    const float* Wqkv   = (const float*)d_in[9];   // [4,256,768]
    const float* bqkv   = (const float*)d_in[10];  // [4,768]
    const float* Wo     = (const float*)d_in[11];  // [4,256,256]
    const float* bo     = (const float*)d_in[12];
    const float* W1     = (const float*)d_in[13];
    const float* b1     = (const float*)d_in[14];
    const float* W2     = (const float*)d_in[15];
    const float* b2     = (const float*)d_in[16];
    const float* ln1_g  = (const float*)d_in[17];
    const float* ln1_b  = (const float*)d_in[18];
    const float* ln2_g  = (const float*)d_in[19];
    const float* ln2_b  = (const float*)d_in[20];

    // workspace layout (floats)
    float* preds = (float*)d_ws;                       // [30,128,256]
    float* qkv   = preds + SEQ_ * BB * DD;             // [128,768]
    float* sc    = qkv   + BB * 3 * DD;                // [4,128,128]
    float* atto  = sc    + 4 * BB * BB;                // [128,256]
    float* t1    = atto  + BB * DD;                    // [128,256]
    float* ff1   = t1    + BB * DD;                    // [128,256]
    float* hA    = ff1   + BB * DD;                    // [128,256]
    float* hB    = hA    + BB * DD;                    // [128,256]

    // ---- encoder: preds[0] = LN(x0 @ enc_W + enc_b) ----
    k_gemm<false, false, true, false><<<dim3(8, 1), 256, 0, stream>>>(
        x, (long)SEQ_ * IMG_, enc_W, DD, enc_b,
        nullptr, 0, encn_g, encn_b,
        preds, DD, IMG_);

    // ---- 29 recurrent steps ----
    for (int t = 1; t < SEQ_; ++t) {
        const float* cur = preds + (size_t)(t - 1) * BB * DD;
        for (int l = 0; l < NL_; ++l) {
            k_gemm<false, false, false, false><<<dim3(8, 6), 256, 0, stream>>>(
                cur, DD, Wqkv + (size_t)l * DD * 3 * DD, 3 * DD,
                bqkv + (size_t)l * 3 * DD,
                nullptr, 0, nullptr, nullptr,
                qkv, 3 * DD, DD);

            k_attn<<<4, 512, 0, stream>>>(qkv, sc, atto);

            // t1 = LN1(cur + atto@Wo + bo)
            k_gemm<true, false, true, false><<<dim3(8, 1), 256, 0, stream>>>(
                atto, DD, Wo + (size_t)l * DD * DD, DD, bo + (size_t)l * DD,
                cur, DD, ln1_g + (size_t)l * DD, ln1_b + (size_t)l * DD,
                t1, DD, DD);

            // ff1 = relu(t1@W1 + b1)
            k_gemm<false, true, false, false><<<dim3(8, 2), 256, 0, stream>>>(
                t1, DD, W1 + (size_t)l * DD * DD, DD, b1 + (size_t)l * DD,
                nullptr, 0, nullptr, nullptr,
                ff1, DD, DD);

            // h = LN2(t1 + ff1@W2 + b2)
            float* outbuf = (l & 1) ? hB : hA;
            k_gemm<true, false, true, false><<<dim3(8, 1), 256, 0, stream>>>(
                ff1, DD, W2 + (size_t)l * DD * DD, DD, b2 + (size_t)l * DD,
                t1, DD, ln2_g + (size_t)l * DD, ln2_b + (size_t)l * DD,
                outbuf, DD, DD);
            cur = outbuf;
        }
        k_layernorm<<<BB, 256, 0, stream>>>(cur, decn_g, decn_b,
                                            preds + (size_t)t * BB * DD);
    }

    // ---- decoder: out[b,t,:] = preds[t,b,:] @ dec_W + dec_b ----
    k_gemm<false, false, false, true><<<dim3((SEQ_ * BB) / 16, IMG_ / 128), 256, 0, stream>>>(
        preds, DD, dec_W, IMG_, dec_b,
        nullptr, 0, nullptr, nullptr,
        (float*)d_out, IMG_, DD);
}
